// WinBlock_26465588478125
// MI455X (gfx1250) — matrix operand order repaired and checked by reading, not compiled, not run on hardware
//
#include <hip/hip_runtime.h>
#include <hip/hip_bf16.h>

typedef __attribute__((ext_vector_type(16))) _Float16 v16h;
typedef __attribute__((ext_vector_type(8)))  float    v8f;

#define C_DIM   384
#define HID_DIM 1536
#define QKV_DIM 1152
#define M_ROWS  100352   // 2048 windows * 49 tokens (divisible by 128)
#define NHEAD   12
#define NWIN    2048

// ---------------- fragment loaders (ISA 7.12.2 layouts) ----------------
// A (16x32 f16): lane<16 -> M=lane, K in {b0..b0+7, b0+16..b0+23}, b0 = 0 (lo) / 8 (hi)
__device__ inline v16h loadA(const _Float16* base, int ld) {
  int lane = threadIdx.x & 31;
  int row  = lane & 15;
  int b0   = (lane < 16) ? 0 : 8;
  union { v16h v; float4 f[2]; } u;
  u.f[0] = *(const float4*)(base + row * ld + b0);
  u.f[1] = *(const float4*)(base + row * ld + b0 + 16);
  return u.v;
}
// B (32x16 f16) from transposed [N][K] tile: lane<16 -> N=lane, K in {b0..b0+7, b0+16..b0+23}, b0 = 0 (lo) / 8 (hi)
__device__ inline v16h loadB(const _Float16* base, int ld) {
  int lane = threadIdx.x & 31;
  int n    = lane & 15;
  int kb   = (lane < 16) ? 0 : 8;
  union { v16h v; float4 f[2]; } u;
  u.f[0] = *(const float4*)(base + n * ld + kb);
  u.f[1] = *(const float4*)(base + n * ld + kb + 16);
  return u.v;
}

__device__ inline size_t win2tok(int row) {  // window-ordered row -> token index
  int win = row / 49, t = row % 49;
  int b = win >> 6, wrem = win & 63;
  int wy = wrem >> 3, wx = wrem & 7;
  int ty = t / 7, tx = t % 7;
  return (size_t)b * 3136 + (size_t)(wy * 7 + ty) * 56 + (wx * 7 + tx);
}
__device__ inline size_t tok2winrow(int token) { // token index -> window-ordered row
  int b = token / 3136, rem = token % 3136;
  int y = rem / 56, xx = rem % 56;
  int wy = y / 7, ty = y % 7, wx = xx / 7, tx = xx % 7;
  return (size_t)(b * 64 + wy * 8 + wx) * 49 + ty * 7 + tx;
}

// ---------------- weight transpose + f32->f16 convert: out[N][K] ----------------
__global__ void transpose_cvt_f16(const float* __restrict__ in, _Float16* __restrict__ out,
                                  int K, int N) {
  int idx = blockIdx.x * 256 + threadIdx.x;
  if (idx < K * N) {
    int k = idx / N, n = idx % N;
    out[(size_t)n * K + k] = (_Float16)in[idx];
  }
}

// ---------------- padded additive attention bias (12 x 64 x 64), mask folded in ----------------
__global__ void build_bias(const float* __restrict__ rel_t, float* __restrict__ biasP) {
  int idx = blockIdx.x * 256 + threadIdx.x;   // 12*4096 entries
  if (idx < NHEAD * 4096) {
    int h = idx >> 12, rem = idx & 4095;
    int m = rem >> 6, nk = rem & 63;
    float v;
    if (nk >= 49)      v = -2e30f;            // key mask (softmax -> 0)
    else if (m >= 49)  v = 0.f;               // padded query rows: harmless
    else {
      int ym = m / 7, xm = m % 7, yk = nk / 7, xk = nk % 7;
      v = rel_t[((ym - yk + 6) * 13 + (xm - xk + 6)) * NHEAD + h];
    }
    biasP[idx] = v;
  }
}

// ---------------- LayerNorm (C=384), optional window re-ordering of output rows ----------------
__global__ void __launch_bounds__(128)
ln_kernel(const float* __restrict__ in, const float* __restrict__ g,
          const float* __restrict__ bta, _Float16* __restrict__ out, int win_order) {
  __shared__ float red1[4], red2[4];
  int token = blockIdx.x;
  int tid = threadIdx.x, lane = tid & 31, w = tid >> 5;
  const float* xr = in + (size_t)token * C_DIM;
  float v[3], s1 = 0.f, s2 = 0.f;
#pragma unroll
  for (int i = 0; i < 3; ++i) { v[i] = xr[tid + i * 128]; s1 += v[i]; s2 += v[i] * v[i]; }
#pragma unroll
  for (int msk = 16; msk >= 1; msk >>= 1) { s1 += __shfl_xor(s1, msk, 32); s2 += __shfl_xor(s2, msk, 32); }
  if (lane == 0) { red1[w] = s1; red2[w] = s2; }
  __syncthreads();
  s1 = red1[0] + red1[1] + red1[2] + red1[3];
  s2 = red2[0] + red2[1] + red2[2] + red2[3];
  float mu = s1 * (1.f / 384.f);
  float var = s2 * (1.f / 384.f) - mu * mu;
  float rs = rsqrtf(var + 1e-5f);
  size_t orow = win_order ? tok2winrow(token) : (size_t)token;
#pragma unroll
  for (int i = 0; i < 3; ++i) {
    int c = tid + i * 128;
    out[orow * C_DIM + c] = (_Float16)((v[i] - mu) * rs * g[c] + bta[c]);
  }
}

// ---------------- generic f16 WMMA GEMM, 128x128 tile, BK=64 (16 WMMAs / barrier) ----------------
// MODE 0: f16 out = v+bias (qkv) | 1: f16 out = gelu(v+bias) (fc1)
// MODE 2: f32 out[win2tok(row)] = v+bias+resid (proj) | 3: f32 out[row] = v+bias+resid (fc2)
template <int MODE>
__global__ void __launch_bounds__(256)
gemm_ws(const _Float16* __restrict__ A, const _Float16* __restrict__ Bt,
        const float* __restrict__ bias, void* __restrict__ outp,
        const float* __restrict__ resid, int M, int N, int K) {
  __shared__ _Float16 As[128 * 64];
  __shared__ _Float16 Bs[128 * 64];
  int m0 = blockIdx.y * 128, n0 = blockIdx.x * 128;
  int tid = threadIdx.x, lane = tid & 31, w = tid >> 5;
  int wm = w >> 1, wn = w & 1;   // 4x2 wave grid, each wave 32x64
  v8f vzero = {};
  v8f acc[2][4];
#pragma unroll
  for (int mi = 0; mi < 2; ++mi)
#pragma unroll
    for (int j = 0; j < 4; ++j) acc[mi][j] = vzero;

  int nk = K >> 6;
  for (int kb = 0; kb < nk; ++kb) {
    int k0 = kb << 6;
#pragma unroll
    for (int i = 0; i < 4; ++i) {       // stage 128x64 A and Bt tiles (float4 = 8 halves)
      int c = tid * 4 + i;              // 0..1023
      int row = c >> 3, off = (c & 7) << 3;
      *(float4*)(As + row * 64 + off) = *(const float4*)(A + (size_t)(m0 + row) * K + k0 + off);
      *(float4*)(Bs + row * 64 + off) = *(const float4*)(Bt + (size_t)(n0 + row) * K + k0 + off);
    }
    if (kb + 1 < nk) {                  // global_prefetch_b8 of next K slab
      __builtin_prefetch(A + (size_t)(m0 + (tid >> 1)) * K + k0 + 64, 0, 1);
      __builtin_prefetch(Bt + (size_t)(n0 + (tid >> 1)) * K + k0 + 64, 0, 1);
    }
    __syncthreads();
#pragma unroll
    for (int kk = 0; kk < 2; ++kk) {
      v16h bfrag[4];
#pragma unroll
      for (int j = 0; j < 4; ++j)
        bfrag[j] = loadB(Bs + (wn * 64 + j * 16) * 64 + kk * 32, 64);
#pragma unroll
      for (int mi = 0; mi < 2; ++mi) {
        v16h afrag = loadA(As + (wm * 32 + mi * 16) * 64 + kk * 32, 64);
#pragma unroll
        for (int j = 0; j < 4; ++j)
          acc[mi][j] = __builtin_amdgcn_wmma_f32_16x16x32_f16(
              false, afrag, false, bfrag[j], (short)0, acc[mi][j], false, false);
      }
    }
    __syncthreads();
  }

#pragma unroll
  for (int mi = 0; mi < 2; ++mi) {
    int rl = wm * 32 + mi * 16 + ((lane < 16) ? 0 : 8);
    size_t rmap[8];                     // hoist row mapping out of j loop
#pragma unroll
    for (int r = 0; r < 8; ++r) {
      int row = m0 + rl + r;
      rmap[r] = (MODE == 2) ? win2tok(row) : (size_t)row;
    }
#pragma unroll
    for (int j = 0; j < 4; ++j) {
      int col = n0 + wn * 64 + j * 16 + (lane & 15);
      float bcol = bias[col];
#pragma unroll
      for (int r = 0; r < 8; ++r) {
        float v = acc[mi][j][r] + bcol;
        if (MODE == 0) {
          ((_Float16*)outp)[rmap[r] * N + col] = (_Float16)v;
        } else if (MODE == 1) {
          v = 0.5f * v * (1.f + erff(v * 0.70710678118654752f));   // exact GELU
          ((_Float16*)outp)[rmap[r] * N + col] = (_Float16)v;
        } else {
          size_t t = rmap[r] * N + col;
          ((float*)outp)[t] = v + resid[t];
        }
      }
    }
  }
}

// ---------------- windowed attention: one 4-wave block per (window, head) ----------------
__global__ void __launch_bounds__(128)
attn_kernel(const _Float16* __restrict__ qkv, const float* __restrict__ biasP,
            _Float16* __restrict__ ao) {
  __shared__ _Float16 Qs[64 * 32];   // query tokens x d   (zero-padded 49->64)
  __shared__ _Float16 Ks[64 * 32];   // key tokens x d     (== B^T tile for S)
  __shared__ _Float16 Vt[32 * 64];   // d x tokens         (== B^T tile for O)
  __shared__ _Float16 Ps[64 * 64];   // softmax probs
  __shared__ float    BiasL[64 * 64];// padded additive bias for this head (mask folded in)
  int bid = blockIdx.x;
  int win = bid / NHEAD, head = bid % NHEAD;
  int tid = threadIdx.x, lane = tid & 31, w = tid >> 5;

  for (int i = tid; i < 2048; i += 128) { Qs[i] = (_Float16)0; Ks[i] = (_Float16)0; Vt[i] = (_Float16)0; }
  {
    const float4* bsrc = (const float4*)(biasP + (size_t)head * 4096);
    float4* bdst = (float4*)BiasL;
#pragma unroll
    for (int i = 0; i < 8; ++i) bdst[tid + i * 128] = bsrc[tid + i * 128];
  }
  __syncthreads();

  const _Float16* base = qkv + (size_t)win * 49 * QKV_DIM + head * 32;
  for (int c = tid; c < 49 * 4; c += 128) {
    int row = c >> 2, off = (c & 3) << 3;
    *(float4*)(Qs + row * 32 + off) = *(const float4*)(base + (size_t)row * QKV_DIM + off);
    *(float4*)(Ks + row * 32 + off) = *(const float4*)(base + 384 + (size_t)row * QKV_DIM + off);
    union { float4 f; _Float16 h[8]; } uv;
    uv.f = *(const float4*)(base + 768 + (size_t)row * QKV_DIM + off);
#pragma unroll
    for (int jj = 0; jj < 8; ++jj) Vt[(off + jj) * 64 + row] = uv.h[jj];
  }
  __syncthreads();

  // S = Q K^T (each wave: 16 rows x 64 cols, one 32-deep WMMA per 16x16 subtile)
  v8f vzero = {};
  v16h aq = loadA(Qs + w * 16 * 32, 32);
  v8f s[4];
#pragma unroll
  for (int j = 0; j < 4; ++j) {
    v16h bk = loadB(Ks + j * 16 * 32, 32);
    s[j] = __builtin_amdgcn_wmma_f32_16x16x32_f16(false, aq, false, bk, (short)0, vzero, false, false);
  }

  // branchless scale + bias(+mask) from LDS
  int mrow = w * 16 + ((lane < 16) ? 0 : 8);
#pragma unroll
  for (int j = 0; j < 4; ++j) {
    int nk2 = j * 16 + (lane & 15);
#pragma unroll
    for (int r = 0; r < 8; ++r)
      s[j][r] = s[j][r] * 0.17677669529663687f + BiasL[(mrow + r) * 64 + nk2];
  }

  // row softmax: each 16-lane half holds a complete 64-wide row per accumulator slot
#pragma unroll
  for (int r = 0; r < 8; ++r) {
    float mx = fmaxf(fmaxf(s[0][r], s[1][r]), fmaxf(s[2][r], s[3][r]));
#pragma unroll
    for (int msk = 8; msk >= 1; msk >>= 1) mx = fmaxf(mx, __shfl_xor(mx, msk, 32));
    float e0 = __expf(s[0][r] - mx), e1 = __expf(s[1][r] - mx);
    float e2 = __expf(s[2][r] - mx), e3 = __expf(s[3][r] - mx);
    float sum = e0 + e1 + e2 + e3;
#pragma unroll
    for (int msk = 8; msk >= 1; msk >>= 1) sum += __shfl_xor(sum, msk, 32);
    float inv = 1.f / sum;
    s[0][r] = e0 * inv; s[1][r] = e1 * inv; s[2][r] = e2 * inv; s[3][r] = e3 * inv;
  }
#pragma unroll
  for (int j = 0; j < 4; ++j) {
    int nk2 = j * 16 + (lane & 15);
#pragma unroll
    for (int r = 0; r < 8; ++r) Ps[(mrow + r) * 64 + nk2] = (_Float16)s[j][r];
  }
  __syncthreads();

  // O = P V  (K-dim 64 -> two 32-deep WMMA steps per 16x16 subtile)
  v8f o0 = vzero, o1 = vzero;
#pragma unroll
  for (int kk = 0; kk < 2; ++kk) {
    v16h ap  = loadA(Ps + w * 16 * 64 + kk * 32, 64);
    v16h bv0 = loadB(Vt + 0 * 64 + kk * 32, 64);
    v16h bv1 = loadB(Vt + 16 * 64 + kk * 32, 64);
    o0 = __builtin_amdgcn_wmma_f32_16x16x32_f16(false, ap, false, bv0, (short)0, o0, false, false);
    o1 = __builtin_amdgcn_wmma_f32_16x16x32_f16(false, ap, false, bv1, (short)0, o1, false, false);
  }
#pragma unroll
  for (int j = 0; j < 2; ++j) {
    int col = head * 32 + j * 16 + (lane & 15);
#pragma unroll
    for (int r = 0; r < 8; ++r) {
      int m = mrow + r;
      float v = (j == 0) ? o0[r] : o1[r];
      if (m < 49) ao[((size_t)win * 49 + m) * C_DIM + col] = (_Float16)v;
    }
  }
}

// ---------------- launch ----------------
extern "C" void kernel_launch(void* const* d_in, const int* in_sizes, int n_in,
                              void* d_out, int out_size, void* d_ws, size_t ws_size,
                              hipStream_t stream) {
  const float* x       = (const float*)d_in[0];
  const float* n1g     = (const float*)d_in[3];
  const float* n1b     = (const float*)d_in[4];
  const float* qkv_w   = (const float*)d_in[5];
  const float* qkv_b   = (const float*)d_in[6];
  const float* proj_w  = (const float*)d_in[7];
  const float* proj_b  = (const float*)d_in[8];
  const float* rel_t   = (const float*)d_in[9];
  const float* n2g     = (const float*)d_in[10];
  const float* n2b     = (const float*)d_in[11];
  const float* fc1_w   = (const float*)d_in[12];
  const float* fc1_b   = (const float*)d_in[13];
  const float* fc2_w   = (const float*)d_in[14];
  const float* fc2_b   = (const float*)d_in[15];

  size_t off = 0;
  auto take = [&](size_t bytes) {
    void* p = (char*)d_ws + off;
    off += (bytes + 255) & ~(size_t)255;
    return p;
  };
  _Float16* Wqkv  = (_Float16*)take((size_t)QKV_DIM * C_DIM * 2);
  _Float16* Wprj  = (_Float16*)take((size_t)C_DIM * C_DIM * 2);
  _Float16* Wfc1  = (_Float16*)take((size_t)HID_DIM * C_DIM * 2);
  _Float16* Wfc2  = (_Float16*)take((size_t)C_DIM * HID_DIM * 2);
  float*    BiasP = (float*)   take((size_t)NHEAD * 4096 * 4);
  _Float16* Xw    = (_Float16*)take((size_t)M_ROWS * C_DIM * 2);
  _Float16* QKV   = (_Float16*)take((size_t)M_ROWS * QKV_DIM * 2);
  _Float16* AO    = (_Float16*)take((size_t)M_ROWS * C_DIM * 2);
  float*    X1    = (float*)   take((size_t)M_ROWS * C_DIM * 4);
  _Float16* Y     = (_Float16*)take((size_t)M_ROWS * C_DIM * 2);
  _Float16* H1    = (_Float16*)take((size_t)M_ROWS * HID_DIM * 2);

  // weight transpose + convert (out: [N][K] f16), padded bias build
  transpose_cvt_f16<<<(C_DIM * QKV_DIM + 255) / 256, 256, 0, stream>>>(qkv_w, Wqkv, C_DIM, QKV_DIM);
  transpose_cvt_f16<<<(C_DIM * C_DIM   + 255) / 256, 256, 0, stream>>>(proj_w, Wprj, C_DIM, C_DIM);
  transpose_cvt_f16<<<(C_DIM * HID_DIM + 255) / 256, 256, 0, stream>>>(fc1_w, Wfc1, C_DIM, HID_DIM);
  transpose_cvt_f16<<<(HID_DIM * C_DIM + 255) / 256, 256, 0, stream>>>(fc2_w, Wfc2, HID_DIM, C_DIM);
  build_bias<<<(NHEAD * 4096 + 255) / 256, 256, 0, stream>>>(rel_t, BiasP);

  // LN1 -> window-ordered f16
  ln_kernel<<<M_ROWS, 128, 0, stream>>>(x, n1g, n1b, Xw, 1);
  // QKV projection
  gemm_ws<0><<<dim3(QKV_DIM / 128, M_ROWS / 128), 256, 0, stream>>>(
      Xw, Wqkv, qkv_b, QKV, nullptr, M_ROWS, QKV_DIM, C_DIM);
  // windowed attention
  attn_kernel<<<NWIN * NHEAD, 128, 0, stream>>>(QKV, BiasP, AO);
  // proj + de-window + residual -> X1 (f32, token order)
  gemm_ws<2><<<dim3(C_DIM / 128, M_ROWS / 128), 256, 0, stream>>>(
      AO, Wprj, proj_b, X1, x, M_ROWS, C_DIM, C_DIM);
  // LN2 -> f16 token order
  ln_kernel<<<M_ROWS, 128, 0, stream>>>(X1, n2g, n2b, Y, 0);
  // fc1 + GELU
  gemm_ws<1><<<dim3(HID_DIM / 128, M_ROWS / 128), 256, 0, stream>>>(
      Y, Wfc1, fc1_b, H1, nullptr, M_ROWS, HID_DIM, C_DIM);
  // fc2 + residual -> d_out
  gemm_ws<3><<<dim3(C_DIM / 128, M_ROWS / 128), 256, 0, stream>>>(
      H1, Wfc2, fc2_b, (float*)d_out, X1, M_ROWS, C_DIM, HID_DIM);
}
